// Linear_489626271767
// MI455X (gfx1250) — matrix operand order repaired and checked by reading, not compiled, not run on hardware
//
#include <hip/hip_runtime.h>

// GEMM: out[M,N] = x[M,K] * w[N,K]^T * scale[N],  M=N=K=4096
// bf16 WMMA (v_wmma_f32_16x16x32_bf16), fp32 accumulate, per-N scale epilogue.
// Roofline: 137 GFLOP, 192 MB mandatory HBM traffic (~8.2 us floor @ 23.3 TB/s);
// AI ~715 flop/B => matrix-core bound, so LDS-staged WMMA is the right path.

typedef __attribute__((ext_vector_type(16))) __bf16        v16bf;
typedef __attribute__((ext_vector_type(4)))  __bf16        bf16x4;
typedef __attribute__((ext_vector_type(8)))  float         v8f;
typedef __attribute__((ext_vector_type(4)))  unsigned int  u32x4;
typedef __attribute__((ext_vector_type(4)))  float         f32x4;
typedef __attribute__((ext_vector_type(4)))  int           i32x4;

#define K_DIM   4096
#define TILE_K  32
#define BLK_M   128
#define BLK_N   128
#define LDS_ROW 40          // halfs per row: 32 data + 8 pad -> 80 bytes (16B aligned)

__device__ __forceinline__ v16bf ld_frag(const unsigned short* p0,
                                         const unsigned short* p1) {
    union { u32x4 q[2]; v16bf v; } u;
    u.q[0] = *(const u32x4*)p0;          // ds_load_b128
    u.q[1] = *(const u32x4*)p1;          // ds_load_b128
    return u.v;
}

__global__ __launch_bounds__(256) void linear_i8w_bf16_wmma(
    const float* __restrict__ X,         // [M, 4096] fp32
    const int*   __restrict__ W,         // [N, 4096] int8-in-int32
    const float* __restrict__ S,         // [N] per-output scale
    float*       __restrict__ O)         // [M, N] fp32
{
    __shared__ __align__(16) unsigned short ldsA[2][BLK_M * LDS_ROW];
    __shared__ __align__(16) unsigned short ldsB[2][BLK_N * LDS_ROW];

    const int tid  = threadIdx.x;
    const int lane = tid & 31;
    const int m0   = blockIdx.y * BLK_M;
    const int n0   = blockIdx.x * BLK_N;

    // wave tile: 32(M) x 64(N) = 2x4 WMMA tiles per wave, 8 waves per block
    const int wave  = tid >> 5;
    const int wm    = (wave & 3) * 32;   // wave M offset in block
    const int wn    = (wave >> 2) * 64;  // wave N offset in block

    // staging assignment: thread covers row = tid/2 (0..127), 16 k-values
    const int srow = tid >> 1;
    const int sseg = (tid & 1) * 16;
    const float* xg = X + (size_t)(m0 + srow) * K_DIM + sseg;
    const int*   wg = W + (size_t)(n0 + srow) * K_DIM + sseg;

    // per-lane fragment addresses (ISA 16-bit A/B layouts)
    const int l15   = lane & 15;
    const int lhalf = lane >> 4;
    // A (16x32): lane=M; half 0 -> k {0-7,16-23}; half 1 -> k {8-15,24-31}
    const int aOff0 = lhalf * 8;         // halfs
    const int aOff1 = 16 + lhalf * 8;
    // B (32x16): lane=N; half 0 -> k {0-7,16-23}; half 1 -> k {8-15,24-31}
    const int bOff0 = lhalf * 8;
    const int bOff1 = 16 + lhalf * 8;

    // per-output-channel scalers for this lane's 4 N-tiles
    float sc[4];
#pragma unroll
    for (int nt = 0; nt < 4; ++nt)
        sc[nt] = S[n0 + wn + nt * 16 + l15];

    v8f acc[2][4];
#pragma unroll
    for (int mt = 0; mt < 2; ++mt)
#pragma unroll
        for (int nt = 0; nt < 4; ++nt)
            acc[mt][nt] = v8f{0.f, 0.f, 0.f, 0.f, 0.f, 0.f, 0.f, 0.f};

    for (int kk = 0, it = 0; kk < K_DIM; kk += TILE_K, ++it) {
        const int buf = it & 1;

        // ---- stage: global fp32/int32 -> bf16 -> LDS ----
        f32x4 xa[4]; i32x4 wa[4];
#pragma unroll
        for (int i = 0; i < 4; ++i) {
            xa[i] = *(const f32x4*)(xg + kk + i * 4);
            wa[i] = *(const i32x4*)(wg + kk + i * 4);
        }
        // prefetch next K-slab (speculative; OOB at tail is dropped)
        __builtin_prefetch(xg + kk + TILE_K, 0, 0);
        __builtin_prefetch(wg + kk + TILE_K, 0, 0);

        // native packed f32->bf16 conversion (v_cvt_pk_bf16_f32 class ops)
        union { bf16x4 h[4]; u32x4 q[2]; } ca, cb;
#pragma unroll
        for (int i = 0; i < 4; ++i) {
            ca.h[i] = __builtin_convertvector(xa[i], bf16x4);
            cb.h[i] = __builtin_convertvector(
                          __builtin_convertvector(wa[i], f32x4), bf16x4);
        }
        unsigned short* la = &ldsA[buf][srow * LDS_ROW + sseg];
        unsigned short* lb = &ldsB[buf][srow * LDS_ROW + sseg];
        *(u32x4*)(la)     = ca.q[0];
        *(u32x4*)(la + 8) = ca.q[1];
        *(u32x4*)(lb)     = cb.q[0];
        *(u32x4*)(lb + 8) = cb.q[1];

        __syncthreads();   // single barrier per step (double-buffered LDS)

        // ---- fragments ----
        v16bf aF[2], bF[4];
#pragma unroll
        for (int mt = 0; mt < 2; ++mt) {
            const unsigned short* base = &ldsA[buf][(wm + mt * 16 + l15) * LDS_ROW];
            aF[mt] = ld_frag(base + aOff0, base + aOff1);
        }
#pragma unroll
        for (int nt = 0; nt < 4; ++nt) {
            const unsigned short* base = &ldsB[buf][(wn + nt * 16 + l15) * LDS_ROW];
            bF[nt] = ld_frag(base + bOff0, base + bOff1);
        }

        // ---- 8 WMMAs ----
#pragma unroll
        for (int mt = 0; mt < 2; ++mt)
#pragma unroll
            for (int nt = 0; nt < 4; ++nt)
                acc[mt][nt] = __builtin_amdgcn_wmma_f32_16x16x32_bf16(
                    /*neg_a=*/false, aF[mt], /*neg_b=*/false, bF[nt],
                    /*c_mod=*/(short)0, acc[mt][nt],
                    /*reuse_a=*/false, /*reuse_b=*/false);
    }

    // ---- epilogue: scale by weight_scaler[n], non-temporal store ----
    // C/D layout: lane = N (lane&15), VGPR v -> M = (lane>>4)*8 + v
    // Output is written once and never re-read: NT keeps x/w resident in L2.
#pragma unroll
    for (int mt = 0; mt < 2; ++mt) {
        const int mbase = m0 + wm + mt * 16 + lhalf * 8;
#pragma unroll
        for (int nt = 0; nt < 4; ++nt) {
            const int n = n0 + wn + nt * 16 + l15;
            const float s = sc[nt];
#pragma unroll
            for (int v = 0; v < 8; ++v)
                __builtin_nontemporal_store(acc[mt][nt][v] * s,
                                            &O[(size_t)(mbase + v) * 4096 + n]);
        }
    }
}

extern "C" void kernel_launch(void* const* d_in, const int* in_sizes, int n_in,
                              void* d_out, int out_size, void* d_ws, size_t ws_size,
                              hipStream_t stream) {
    const float* X = (const float*)d_in[0];   // x [B,S,4096] fp32
    const int*   W = (const int*)d_in[1];     // weight [4096,4096] int32 (int8 values)
    const float* S = (const float*)d_in[2];   // weight_scaler [4096] fp32
    float*       O = (float*)d_out;           // out [B,S,4096] fp32

    const int N = in_sizes[2];                // 4096
    const int M = in_sizes[0] / K_DIM;        // B*S = 4096

    dim3 grid(N / BLK_N, M / BLK_M);          // 32 x 32 blocks
    linear_i8w_bf16_wmma<<<grid, 256, 0, stream>>>(X, W, S, O);
}
